// WordEmbedding_12790412608052
// MI455X (gfx1250) — compile-verified, matrix operand order checked by reading, not run on hardware
//
#include <hip/hip_runtime.h>
#include <hip/hip_bf16.h>
#include <math.h>

// Problem constants (from reference)
#define B_SZ   8192
#define LMAXC  12
#define TOUT   10
#define EMB    300
#define EMBP   320          // padded K for WMMA (multiple of 64)
#define CVEC   256
#define HID    256
#define GATES  1024         // 4*HID
#define OUTD   512
#define ROWS   (B_SZ * TOUT)   // 81920

// GEMM tiling
#define TM 128   // block M tile (8 waves x 16)
#define TN 64    // block N tile
#define TK 64    // K chunk staged in LDS

typedef __bf16 bf16;
typedef __attribute__((ext_vector_type(8)))  __bf16 v8bf;
typedef __attribute__((ext_vector_type(16))) __bf16 v16bf;
typedef __attribute__((ext_vector_type(8)))  float  v8f;

// ---------------------------------------------------------------------------
// CDNA5 async copy: global -> LDS, 16 bytes per lane, tracked by ASYNCcnt.
// GV addressing: vdst = LDS byte offset VGPR, vaddr = 64-bit global address.
// (Low 32 bits of a generic pointer to __shared__ are the LDS byte address.)
// ---------------------------------------------------------------------------
__device__ __forceinline__ void async_copy_b128(const bf16* gsrc, bf16* ldst) {
  unsigned loff = (unsigned)(size_t)ldst;
  asm volatile("global_load_async_to_lds_b128 %0, %1, off"
               :: "v"(loff), "v"(gsrc)
               : "memory");
}
#define WAIT_ASYNCCNT(n) asm volatile("s_wait_asynccnt %0" :: "i"(n) : "memory")

// Load one 16x32 bf16 WMMA fragment row for this lane: two contiguous 16B
// chunks. Layout per ISA: lanes 0-15 hold K[0..7],K[16..23]; lanes 16-31 hold
// K[8..15],K[24..31].
__device__ __forceinline__ v16bf load_frag16(const bf16* __restrict__ p, int half) {
  v8bf lo = *(const v8bf*)(p + half * 8);
  v8bf hi = *(const v8bf*)(p + 16 + half * 8);
  return __builtin_shufflevector(lo, hi, 0,1,2,3,4,5,6,7,8,9,10,11,12,13,14,15);
}

// ---------------------------------------------------------------------------
// Async-LDS double-buffered WMMA GEMM: out = act(A @ W^T + bias + bias2)
//   A: (M, lda) bf16 row-major, W: (N, ldw) bf16 row-major, ldc == N.
//   Block = 256 threads (8 waves). Block tile 128(M) x 64(N); wave tile 16x64.
//   M % 128 == 0, N % 64 == 0, K % 64 == 0 (guaranteed by caller).
// ---------------------------------------------------------------------------
__global__ void __launch_bounds__(256)
gemm_wmma_bf16(const bf16* __restrict__ A, int lda,
               const bf16* __restrict__ W, int ldw,
               const float* __restrict__ bias,
               const float* __restrict__ bias2,
               float* __restrict__ outF,
               bf16* __restrict__ outB,
               int N, int K, int relu)
{
  __shared__ __align__(16) bf16 sA[2][TM * TK];   // 2 x 16 KB
  __shared__ __align__(16) bf16 sW[2][TN * TK];   // 2 x  8 KB

  const int tid  = threadIdx.x;
  const int lane = tid & 31;
  const int wave = tid >> 5;
  const int half = lane >> 4;     // which 16-lane group
  const int l16  = lane & 15;
  const long m0  = (long)blockIdx.y * TM;
  const int  n0  = blockIdx.x * TN;

  // Stage loader: 128x64 A tile = 1024 16B-chunks (4/thread),
  //               64x64  W tile =  512 16B-chunks (2/thread).
  const bf16* __restrict__ Ab = A + m0 * (long)lda;
  const bf16* __restrict__ Wb = W + (long)n0 * (long)ldw;

  auto load_stage = [&](int buf, int k0) {
    #pragma unroll
    for (int i = 0; i < 4; ++i) {
      int c = tid + i * 256;
      int r = c >> 3, col = (c & 7) * 8;          // col in elements (16B chunk)
      async_copy_b128(Ab + (long)r * lda + k0 + col, &sA[buf][r * TK + col]);
    }
    #pragma unroll
    for (int i = 0; i < 2; ++i) {
      int c = tid + i * 256;
      int r = c >> 3, col = (c & 7) * 8;
      async_copy_b128(Wb + (long)r * ldw + k0 + col, &sW[buf][r * TK + col]);
    }
  };

  v8f acc[4];
  #pragma unroll
  for (int j = 0; j < 4; ++j) acc[j] = v8f{};

  const int nk = K / TK;
  load_stage(0, 0);                               // prologue: stage 0 in flight

  for (int kk = 0; kk < nk; ++kk) {
    const int buf = kk & 1;
    if (kk + 1 < nk) {
      load_stage(buf ^ 1, (kk + 1) * TK);         // prefetch next stage
      WAIT_ASYNCCNT(6);                           // stage kk done, next in flight
    } else {
      WAIT_ASYNCCNT(0);
    }
    __syncthreads();                              // tile visible to all waves

    const bf16* sa = &sA[buf][(wave * 16 + l16) * TK];
    const bf16* sw = &sW[buf][l16 * TK];
    #pragma unroll
    for (int ks = 0; ks < TK; ks += 32) {
      v16bf af = load_frag16(sa + ks, half);
      #pragma unroll
      for (int j = 0; j < 4; ++j) {
        v16bf wf = load_frag16(sw + j * 16 * TK + ks, half);
        acc[j] = __builtin_amdgcn_wmma_f32_16x16x32_bf16(
                   /*neg_a=*/false, af, /*neg_b=*/false, wf,
                   /*c_mod=*/(short)0, acc[j],
                   /*reuse_a=*/false, /*reuse_b=*/false);
      }
    }
    __syncthreads();                              // done reading before reuse
  }

  // C/D layout: lane(l16) = N column, VGPR v -> M = half*8 + v
  #pragma unroll
  for (int j = 0; j < 4; ++j) {
    const int n = n0 + j * 16 + l16;
    float badd = (bias  ? bias[n]  : 0.0f) + (bias2 ? bias2[n] : 0.0f);
    #pragma unroll
    for (int v = 0; v < 8; ++v) {
      long m = m0 + wave * 16 + half * 8 + v;
      float x = acc[j][v] + badd;
      if (relu) x = fmaxf(x, 0.0f);
      long idx = m * (long)N + n;
      if (outF) outF[idx] = x;
      if (outB) outB[idx] = (bf16)x;
    }
  }
}

// ---------------------------------------------------------------------------
// Gather + linear interpolation along char axis, write padded bf16 activation.
// embp: (ROWS, EMBP) bf16, cols [EMB..EMBP) zero.
// ---------------------------------------------------------------------------
__global__ void interp_embed_kernel(const int* __restrict__ words,
                                    const int* __restrict__ lengths,
                                    const float* __restrict__ emb_table,
                                    bf16* __restrict__ embp)
{
  long gid = (long)blockIdx.x * blockDim.x + threadIdx.x;
  if (gid >= (long)ROWS * EMBP) return;
  int k   = (int)(gid % EMBP);
  int row = (int)(gid / EMBP);
  if (k >= EMB) { embp[gid] = (bf16)0.0f; return; }
  int b = row / TOUT, t = row % TOUT;
  int L = lengths[b];
  float Lf  = (float)(L - 1);
  float pos = (float)t * Lf / 9.0f;
  int i0 = (int)floorf(pos);
  if (i0 > L - 1) i0 = L - 1;
  int i1 = i0 + 1; if (i1 > L - 1) i1 = L - 1;
  float f = pos - (float)i0;
  int w0 = words[b * LMAXC + i0];
  int w1 = words[b * LMAXC + i1];
  float v = emb_table[(long)w0 * EMB + k] * (1.0f - f)
          + emb_table[(long)w1 * EMB + k] * f;
  embp[gid] = (bf16)v;
}

// f32 (N, Ksrc) -> bf16 (N, Kp), zero-padding K.
__global__ void convert_pad_kernel(const float* __restrict__ src,
                                   bf16* __restrict__ dst,
                                   int N, int Ksrc, int Kp)
{
  long gid = (long)blockIdx.x * blockDim.x + threadIdx.x;
  if (gid >= (long)N * Kp) return;
  int k = (int)(gid % Kp);
  int n = (int)(gid / Kp);
  dst[gid] = (k < Ksrc) ? (bf16)src[(long)n * Ksrc + k] : (bf16)0.0f;
}

// Concatenate [Wih | Whh] (each (1024,256) f32) -> (1024,512) bf16.
__global__ void concat_w_kernel(const float* __restrict__ wih,
                                const float* __restrict__ whh,
                                bf16* __restrict__ wcat)
{
  long gid = (long)blockIdx.x * blockDim.x + threadIdx.x;
  if (gid >= (long)GATES * 512) return;
  int k = (int)(gid & 511);
  int n = (int)(gid >> 9);
  float v = (k < 256) ? wih[(long)n * 256 + k] : whh[(long)n * 256 + (k - 256)];
  wcat[gid] = (bf16)v;
}

__global__ void init_state_kernel(bf16* __restrict__ h, float* __restrict__ c)
{
  long gid = (long)blockIdx.x * blockDim.x + threadIdx.x;
  if (gid >= (long)B_SZ * HID) return;
  h[gid] = (bf16)0.0f;
  c[gid] = 0.0f;
}

// Build per-step activation [x_t | h] : (B, 512) bf16.
__global__ void build_acat_kernel(const bf16* __restrict__ cv,
                                  const bf16* __restrict__ h,
                                  bf16* __restrict__ acat, int t)
{
  long gid = (long)blockIdx.x * blockDim.x + threadIdx.x;
  if (gid >= (long)B_SZ * 512) return;
  int j = (int)(gid & 511);
  int b = (int)(gid >> 9);
  acat[gid] = (j < 256) ? cv[((long)b * TOUT + t) * CVEC + j]
                        : h[(long)b * HID + (j - 256)];
}

// LSTM cell: gate order i,f,g,o. Writes new c (f32), h (bf16), and rec slice.
__global__ void lstm_cell_kernel(const float* __restrict__ g,
                                 float* __restrict__ c,
                                 bf16* __restrict__ h,
                                 bf16* __restrict__ rec,
                                 int t, int rec_off)
{
  long gid = (long)blockIdx.x * blockDim.x + threadIdx.x;
  if (gid >= (long)B_SZ * HID) return;
  int b = (int)(gid >> 8);
  int j = (int)(gid & 255);
  const float* gr = g + (long)b * GATES;
  float gi = gr[j], gf = gr[256 + j], gg = gr[512 + j], go = gr[768 + j];
  float si = 1.0f / (1.0f + expf(-gi));
  float sf = 1.0f / (1.0f + expf(-gf));
  float so = 1.0f / (1.0f + expf(-go));
  float cn = sf * c[gid] + si * tanhf(gg);
  float hn = so * tanhf(cn);
  c[gid] = cn;
  h[gid] = (bf16)hn;
  rec[((long)b * TOUT + t) * 512 + rec_off + j] = (bf16)hn;
}

// ---------------------------------------------------------------------------
extern "C" void kernel_launch(void* const* d_in, const int* in_sizes, int n_in,
                              void* d_out, int out_size, void* d_ws, size_t ws_size,
                              hipStream_t stream)
{
  const int*   words   = (const int*)  d_in[0];
  const int*   lengths = (const int*)  d_in[1];
  const float* emb_t   = (const float*)d_in[2];
  const float* enc_w   = (const float*)d_in[3];
  const float* enc_b   = (const float*)d_in[4];
  const float* wih_f   = (const float*)d_in[5];
  const float* whh_f   = (const float*)d_in[6];
  const float* bih_f   = (const float*)d_in[7];
  const float* bhh_f   = (const float*)d_in[8];
  const float* wih_b   = (const float*)d_in[9];
  const float* whh_b   = (const float*)d_in[10];
  const float* bih_b   = (const float*)d_in[11];
  const float* bhh_b   = (const float*)d_in[12];
  const float* out_w   = (const float*)d_in[13];
  const float* out_b   = (const float*)d_in[14];
  float* out = (float*)d_out;
  (void)in_sizes; (void)n_in; (void)out_size; (void)ws_size;

  // Workspace carve (256B aligned so b128 loads are legal).
  char* p = (char*)d_ws;
  auto carve = [&](size_t bytes) -> void* {
    void* r = (void*)p;
    p += (bytes + 255) & ~(size_t)255;
    return r;
  };
  bf16*  embp   = (bf16*) carve((size_t)ROWS * EMBP * 2);   // interp activations
  bf16*  encw_b = (bf16*) carve((size_t)CVEC * EMBP * 2);   // encoder weight
  bf16*  cv     = (bf16*) carve((size_t)ROWS * CVEC * 2);   // encoder output
  bf16*  wcat_f = (bf16*) carve((size_t)GATES * 512 * 2);   // [Wih|Whh] fwd
  bf16*  wcat_b = (bf16*) carve((size_t)GATES * 512 * 2);   // [Wih|Whh] bwd
  bf16*  outw_b = (bf16*) carve((size_t)OUTD * 512 * 2);    // output weight
  bf16*  rec    = (bf16*) carve((size_t)ROWS * 512 * 2);    // [h_f|h_b]
  bf16*  acat   = (bf16*) carve((size_t)B_SZ * 512 * 2);    // per-step [x|h]
  bf16*  hbuf   = (bf16*) carve((size_t)B_SZ * HID * 2);
  float* cbuf   = (float*)carve((size_t)B_SZ * HID * 4);
  float* gtmp   = (float*)carve((size_t)B_SZ * GATES * 4);  // gate pre-activations

  // 1) gather + interp -> padded bf16
  {
    long total = (long)ROWS * EMBP;
    interp_embed_kernel<<<(unsigned)((total + 255) / 256), 256, 0, stream>>>(
        words, lengths, emb_t, embp);
  }

  // 2) weight conversions
  convert_pad_kernel<<<(unsigned)(((long)CVEC * EMBP + 255) / 256), 256, 0, stream>>>(
      enc_w, encw_b, CVEC, EMB, EMBP);
  concat_w_kernel<<<(unsigned)(((long)GATES * 512 + 255) / 256), 256, 0, stream>>>(
      wih_f, whh_f, wcat_f);
  concat_w_kernel<<<(unsigned)(((long)GATES * 512 + 255) / 256), 256, 0, stream>>>(
      wih_b, whh_b, wcat_b);
  convert_pad_kernel<<<(unsigned)(((long)OUTD * 512 + 255) / 256), 256, 0, stream>>>(
      out_w, outw_b, OUTD, 512, 512);

  // 3) encoder GEMM: cv = relu(embp @ enc_w^T + enc_b)
  {
    dim3 grid(CVEC / TN, ROWS / TM);
    gemm_wmma_bf16<<<grid, 256, 0, stream>>>(embp, EMBP, encw_b, EMBP,
                                             enc_b, nullptr, nullptr, cv,
                                             CVEC, EMBP, /*relu=*/1);
  }

  // 4) bidirectional LSTM, one fused GEMM per step
  for (int dir = 0; dir < 2; ++dir) {
    const bf16*  wcat = (dir == 0) ? wcat_f : wcat_b;
    const float* bih  = (dir == 0) ? bih_f  : bih_b;
    const float* bhh  = (dir == 0) ? bhh_f  : bhh_b;
    const int rec_off = (dir == 0) ? 0 : 256;
    init_state_kernel<<<(unsigned)(((long)B_SZ * HID + 255) / 256), 256, 0, stream>>>(
        hbuf, cbuf);
    for (int s = 0; s < TOUT; ++s) {
      int t = (dir == 0) ? s : (TOUT - 1 - s);
      build_acat_kernel<<<(unsigned)(((long)B_SZ * 512 + 255) / 256), 256, 0, stream>>>(
          cv, hbuf, acat, t);
      dim3 grid(GATES / TN, B_SZ / TM);
      gemm_wmma_bf16<<<grid, 256, 0, stream>>>(acat, 512, wcat, 512,
                                               bih, bhh, gtmp, nullptr,
                                               GATES, 512, /*relu=*/0);
      lstm_cell_kernel<<<(unsigned)(((long)B_SZ * HID + 255) / 256), 256, 0, stream>>>(
          gtmp, cbuf, hbuf, rec, t, rec_off);
    }
  }

  // 5) output GEMM: out = rec @ out_w^T + out_b   (f32 output)
  {
    dim3 grid(OUTD / TN, ROWS / TM);
    gemm_wmma_bf16<<<grid, 256, 0, stream>>>(rec, 512, outw_b, 512,
                                             out_b, nullptr, out, nullptr,
                                             OUTD, 512, /*relu=*/0);
  }
}
